// LocalSelfAttention2d_22797686408230
// MI455X (gfx1250) — matrix operand order repaired and checked by reading, not compiled, not run on hardware
//
#include <hip/hip_runtime.h>
#include <hip/hip_bf16.h>

// ---------------------------------------------------------------------------
// Local windowed self-attention (P=8, 8 heads, dh=32) for MI455X / gfx1250.
// Pre-pass: pad+convert x -> fp16 (pad baked in), convert weights -> fp16.
// Pipeline: [qkv GEMM + bias + pos + scale] -> [per-window attention] -> [out GEMM].
// Matmuls: v_wmma_f32_16x16x32_f16 (wave32 WMMA), f32 accumulation.
// Staging: global_load_async_to_lds_b128 (ASYNCcnt) for all pure fp16 copies;
// phase-split vector loads for the one transpose-scatter (x tile, V tile).
// ---------------------------------------------------------------------------

typedef __attribute__((ext_vector_type(16))) _Float16 v16h;
typedef __attribute__((ext_vector_type(8)))  _Float16 v8h;
typedef __attribute__((ext_vector_type(8)))  float    v8f;

#define NBATCH 4
#define CIN    256
#define HC     256
#define HEADS  8
#define DH     32
#define PW     8
#define HIMG   128
#define WIMG   128
#define HPAD   136
#define WPAD   136
#define NH     17
#define NWIN   289      // 17*17
#define PADT   4        // pad_h//2 == pad_w//2 == 4
#define QSCALE 0.17677669529663687f  // 1/sqrt(32)

// D = A(16x32 f16) * B(32x16 f16) + C(16x16 f32)
__device__ __forceinline__ v8f wmma16(v16h a, v16h b, v8f c) {
  return __builtin_amdgcn_wmma_f32_16x16x32_f16(false, a, false, b, (short)0, c,
                                                false, false);
}

// Per-lane async copy of 16 bytes global -> LDS, tracked by ASYNCcnt.
__device__ __forceinline__ void async_b128(void* lds_dst, const void* gbase,
                                           unsigned byte_off) {
  const unsigned lds_addr = (unsigned)(size_t)lds_dst;  // LDS aperture: low 32b
  asm volatile("global_load_async_to_lds_b128 %0, %1, %2"
               :
               : "v"(lds_addr), "v"(byte_off), "s"((unsigned long long)(size_t)gbase)
               : "memory");
}

__device__ __forceinline__ void wait_async() {
  asm volatile("s_wait_asynccnt 0x0" ::: "memory");
}

// A-matrix fragment (16x32, f16) from a row-major LDS tile.
// ISA layout: lane<16 holds row M=lane, K = {8g+0..7, 16+8g+0..7}, g = lane>=16.
__device__ __forceinline__ v16h load_a_frag(const _Float16* base, int lane,
                                            int rowStride) {
  const int m = lane & 15;
  const int g = (lane >> 4) & 1;
  const _Float16* row = base + m * rowStride;
  v8h lo = *(const v8h*)(row + g * 8);
  v8h hi = *(const v8h*)(row + 16 + g * 8);
  v16h a;
#pragma unroll
  for (int j = 0; j < 8; ++j) { a[j] = lo[j]; a[j + 8] = hi[j]; }
  return a;
}

// B-matrix fragment (32x16, f16). LDS tile stored "n-major": baseT[n][kk].
// Lane<16: n=lane, kk = {8g+0..7, 16+8g+0..7} (matches A fragment).
__device__ __forceinline__ v16h load_b_frag(const _Float16* baseT, int lane,
                                            int rowStride) {
  const int n = lane & 15;
  const int g = (lane >> 4) & 1;
  const _Float16* row = baseT + n * rowStride;
  v8h lo = *(const v8h*)(row + g * 8);
  v8h hi = *(const v8h*)(row + 16 + g * 8);
  v16h b;
#pragma unroll
  for (int j = 0; j < 8; ++j) { b[j] = lo[j]; b[j + 8] = hi[j]; }
  return b;
}

// ---------------------------------------------------------------------------
// Pre-pass: pad + convert x[B][C][128][128] f32 -> xh[B][C][136][136] f16.
// ---------------------------------------------------------------------------
__global__ __launch_bounds__(256) void padcvt_kernel(
    const float* __restrict__ x, _Float16* __restrict__ xh) {
  const int i = blockIdx.x * 256 + threadIdx.x;
  const int total = NBATCH * CIN * HPAD * WPAD;
  if (i >= total) return;
  const int wp = i % WPAD;
  const int t1 = i / WPAD;
  const int hp = t1 % HPAD;
  const int bc = t1 / HPAD;
  const int h = hp - PADT, w = wp - PADT;
  float v = 0.0f;
  if ((unsigned)h < (unsigned)HIMG && (unsigned)w < (unsigned)WIMG)
    v = x[((size_t)bc * HIMG + h) * WIMG + w];
  xh[i] = (_Float16)v;
}

// Pre-pass: f32 -> f16 elementwise (weights).
__global__ __launch_bounds__(256) void cvt_kernel(const float* __restrict__ s,
                                                  _Float16* __restrict__ d,
                                                  int n) {
  const int i = blockIdx.x * 256 + threadIdx.x;
  if (i < n) d[i] = (_Float16)s[i];
}

// ---------------------------------------------------------------------------
// Kernel 1: QKV projection. One block per (window, 128-channel tile of 768).
// A tile: one v8h = one window row per channel (aligned, unconditional) +
// LDS transpose scatter, phase-split. B tile: async fp16 copy.
// Outputs fp16 q/k/v in windowed layout [b*8+head][win][64][32]; bias,
// position and 1/sqrt(dh) folded into the q epilogue.
// ---------------------------------------------------------------------------
__global__ __launch_bounds__(128) void qkv_kernel(
    const _Float16* __restrict__ xh, const _Float16* __restrict__ whq,
    const float* __restrict__ b_qkv, const float* __restrict__ position,
    _Float16* __restrict__ qb, _Float16* __restrict__ kb,
    _Float16* __restrict__ vb) {
  __shared__ __attribute__((aligned(32))) _Float16 Als[64 * 32];    // [pix][k]
  __shared__ __attribute__((aligned(32))) _Float16 Bls[128 * 32];   // [n][k]

  const int tid  = threadIdx.x;
  const int lane = tid & 31;
  const int wv   = tid >> 5;

  const int tile = blockIdx.x;          // 0..1155 = b*289 + win
  const int b    = tile / NWIN;
  const int win  = tile % NWIN;
  const int wy   = win / NH;
  const int wx   = win % NH;
  const int o0   = blockIdx.y * 128;    // 0..640

  v8f acc[8] = {};

  for (int kc = 0; kc < CIN / 32; ++kc) {
    __syncthreads();
    // B tile: pure fp16 copy -> async to LDS (4 x b128 per lane).
#pragma unroll
    for (int it = 0; it < 4; ++it) {
      const int e = (tid + it * 128) * 8;   // half index in 128x32 tile
      const int n = e >> 5, j = e & 31;
      async_b128(&Bls[e], whq,
                 (unsigned)(((o0 + n) * CIN + kc * 32 + j) * sizeof(_Float16)));
    }
    // A tile: v8h row loads (8 pixels x 1 channel), phase-split, then scatter.
    v8h rowv[2];
#pragma unroll
    for (int it = 0; it < 2; ++it) {
      const int idx = tid + it * 128;       // 0..255
      const int j   = idx >> 3;             // channel 0..31
      const int py  = idx & 7;              // window row
      rowv[it] = *(const v8h*)&xh[(((size_t)b * CIN + kc * 32 + j) * HPAD +
                                   wy * PW + py) * WPAD + wx * PW];
    }
#pragma unroll
    for (int it = 0; it < 2; ++it) {
      const int idx = tid + it * 128;
      const int j   = idx >> 3;
      const int py  = idx & 7;
#pragma unroll
      for (int px = 0; px < 8; ++px)
        Als[(py * 8 + px) * 32 + j] = rowv[it][px];
    }
    wait_async();
    __syncthreads();

    const v16h a = load_a_frag(&Als[wv * 16 * 32], lane, 32);
#pragma unroll
    for (int t = 0; t < 8; ++t) {
      const v16h bf = load_b_frag(&Bls[t * 16 * 32], lane, 32);
      acc[t] = wmma16(a, bf, acc[t]);
    }
  }

  // Epilogue: scatter C tiles into windowed q/k/v (fp16).
  const int n_ = lane & 15;
  const int g  = (lane >> 4) & 1;
#pragma unroll
  for (int t = 0; t < 8; ++t) {
    const int ochan = o0 + t * 16 + n_;
    const int which = ochan >> 8;            // 0=q 1=k 2=v
    const int hcch  = ochan & 255;
    const int head  = hcch >> 5;
    const int d     = hcch & 31;
    _Float16* dst = (which == 0) ? qb : (which == 1) ? kb : vb;
    const size_t base = ((size_t)(b * HEADS + head) * NWIN + win) * (64 * DH);
    const float bias = b_qkv[ochan];
#pragma unroll
    for (int r = 0; r < 8; ++r) {
      const int p = wv * 16 + r + 8 * g;     // window pixel 0..63
      float val = acc[t][r] + bias;
      if (which == 0)
        val = (val + position[hcch * 64 + p]) * QSCALE;  // pos + 1/sqrt(dh)
      dst[base + (size_t)p * DH + d] = (_Float16)val;
    }
  }
}

// ---------------------------------------------------------------------------
// Kernel 2: attention. One block per (b, head, window); 4 waves, each owns a
// 16-row strip of Q. S = Q*K^T (4 WMMAs), row softmax in LDS, O = P*V
// (4 WMMAs). Q/K staged async; V staged transposed (phase-split).
// Output fp16 in pixel-major layout [b][win][p][256].
// ---------------------------------------------------------------------------
__global__ __launch_bounds__(128) void attn_kernel(
    const _Float16* __restrict__ qb, const _Float16* __restrict__ kb,
    const _Float16* __restrict__ vb, _Float16* __restrict__ ob) {
  __shared__ __attribute__((aligned(32))) _Float16 Qls[64 * 32];
  __shared__ __attribute__((aligned(32))) _Float16 Kls[64 * 32];
  __shared__ __attribute__((aligned(32))) _Float16 VT[32 * 64];  // transposed
  __shared__ float Sls[4][16 * 64];
  __shared__ __attribute__((aligned(32))) _Float16 Pls[4][16 * 64];

  const int tid  = threadIdx.x;
  const int lane = tid & 31;
  const int wv   = tid >> 5;

  const int bid  = blockIdx.x;           // bh*289 + win
  const int bh   = bid / NWIN;
  const int win  = bid % NWIN;
  const int b    = bh >> 3;
  const int head = bh & 7;
  const size_t base = (size_t)bid * (64 * DH);

  // Async-stage Q and K: 4 KB each, 16 B per lane per issue (ASYNCcnt).
#pragma unroll
  for (int it = 0; it < 2; ++it) {
    const int e = (tid + it * 128) * 8;  // half-index, 8 halfs per b128
    const unsigned go = (unsigned)((base + e) * sizeof(_Float16));
    async_b128(&Qls[e], qb, go);
    async_b128(&Kls[e], kb, go);
  }
  // V staged transposed (VT[dh][pos]): phase-split v8h loads + scatter.
  {
    v8h vv[2];
#pragma unroll
    for (int it = 0; it < 2; ++it) {
      const int e = (tid + it * 128) * 8;
      vv[it] = *(const v8h*)&vb[base + e];
    }
#pragma unroll
    for (int it = 0; it < 2; ++it) {
      const int e = (tid + it * 128) * 8;
      const int p = e >> 5, d0 = e & 31;
#pragma unroll
      for (int k = 0; k < 8; ++k) VT[(d0 + k) * 64 + p] = vv[it][k];
    }
  }
  wait_async();
  __syncthreads();

  // S strip: 16 rows x 64 cols of logits (scale already folded into q).
  const v16h qa = load_a_frag(&Qls[wv * 16 * 32], lane, 32);
  const int n_ = lane & 15;
  const int g  = (lane >> 4) & 1;
#pragma unroll
  for (int t = 0; t < 4; ++t) {
    v8f s = {};
    const v16h kf = load_b_frag(&Kls[t * 16 * 32], lane, 32);  // B = K^T
    s = wmma16(qa, kf, s);
#pragma unroll
    for (int r = 0; r < 8; ++r)
      Sls[wv][(r + 8 * g) * 64 + t * 16 + n_] = s[r];
  }
  __syncthreads();

  // Row softmax (lanes 0..15 each own one strip row); P stored as f16.
  if (lane < 16) {
    float* row = &Sls[wv][lane * 64];
    float mx = row[0];
    for (int j = 1; j < 64; ++j) mx = fmaxf(mx, row[j]);
    float sum = 0.0f;
    for (int j = 0; j < 64; ++j) { float e = __expf(row[j] - mx); row[j] = e; sum += e; }
    const float inv = 1.0f / sum;
    _Float16* prow = &Pls[wv][lane * 64];
    for (int j = 0; j < 64; ++j) prow[j] = (_Float16)(row[j] * inv);
  }
  __syncthreads();

  // O strip = P(16x64) * V(64x32): 2 n-tiles x 2 K-chunks of 32.
#pragma unroll
  for (int nt = 0; nt < 2; ++nt) {
    v8f oc = {};
#pragma unroll
    for (int kcc = 0; kcc < 2; ++kcc) {
      const v16h pa = load_a_frag(&Pls[wv][kcc * 32], lane, 64);
      const v16h vf = load_b_frag(&VT[(nt * 16) * 64 + kcc * 32], lane, 64);
      oc = wmma16(pa, vf, oc);
    }
#pragma unroll
    for (int r = 0; r < 8; ++r) {
      const int p = wv * 16 + r + 8 * g;
      const int d = nt * 16 + n_;
      ob[((size_t)(b * NWIN + win) * 64 + p) * HC + head * DH + d] =
          (_Float16)oc[r];
    }
  }
}

// ---------------------------------------------------------------------------
// Kernel 3: output projection GEMM + crop. One block per (window, 128-channel
// tile of 256). Both tiles staged with async-to-LDS. y[b][o][h][w] fp32.
// ---------------------------------------------------------------------------
__global__ __launch_bounds__(128) void out_kernel(
    const _Float16* __restrict__ ob, const _Float16* __restrict__ who,
    const float* __restrict__ b_out, float* __restrict__ y) {
  __shared__ __attribute__((aligned(32))) _Float16 Als[64 * 32];
  __shared__ __attribute__((aligned(32))) _Float16 Bls[128 * 32];

  const int tid  = threadIdx.x;
  const int lane = tid & 31;
  const int wv   = tid >> 5;

  const int tile = blockIdx.x;
  const int b    = tile / NWIN;
  const int win  = tile % NWIN;
  const int wy   = win / NH;
  const int wx   = win % NH;
  const int o0   = blockIdx.y * 128;    // 0 or 128
  const size_t rb = (size_t)(b * NWIN + win) * 64 * HC;

  v8f acc[8] = {};

  for (int kc = 0; kc < HC / 32; ++kc) {
    __syncthreads();
    // A tile: async fp16 copy from pixel-major o buffer.
#pragma unroll
    for (int it = 0; it < 2; ++it) {
      const int e = (tid + it * 128) * 8;   // half-index within 64x32 tile
      const int p = e >> 5, j = e & 31;
      async_b128(&Als[e], ob,
                 (unsigned)((rb + (size_t)p * HC + kc * 32 + j) * sizeof(_Float16)));
    }
    // B tile: async fp16 copy from converted weights.
#pragma unroll
    for (int it = 0; it < 4; ++it) {
      const int e = (tid + it * 128) * 8;   // half index in 128x32 tile
      const int n = e >> 5, j = e & 31;
      async_b128(&Bls[e], who,
                 (unsigned)(((o0 + n) * HC + kc * 32 + j) * sizeof(_Float16)));
    }
    wait_async();
    __syncthreads();

    const v16h a = load_a_frag(&Als[wv * 16 * 32], lane, 32);
#pragma unroll
    for (int t = 0; t < 8; ++t) {
      const v16h bf = load_b_frag(&Bls[t * 16 * 32], lane, 32);
      acc[t] = wmma16(a, bf, acc[t]);
    }
  }

  const int n_ = lane & 15;
  const int g  = (lane >> 4) & 1;
#pragma unroll
  for (int t = 0; t < 8; ++t) {
    const int o = o0 + t * 16 + n_;
    const float bias = b_out[o];
#pragma unroll
    for (int r = 0; r < 8; ++r) {
      const int p  = wv * 16 + r + 8 * g;
      const int hp = wy * PW + (p >> 3), wp = wx * PW + (p & 7);
      const int h = hp - PADT, w = wp - PADT;
      if ((unsigned)h < (unsigned)HIMG && (unsigned)w < (unsigned)WIMG)
        y[(((size_t)b * CIN + o) * HIMG + h) * WIMG + w] = acc[t][r] + bias;
    }
  }
}

// ---------------------------------------------------------------------------
extern "C" void kernel_launch(void* const* d_in, const int* in_sizes, int n_in,
                              void* d_out, int out_size, void* d_ws,
                              size_t ws_size, hipStream_t stream) {
  const float* x        = (const float*)d_in[0];
  const float* w_qkv    = (const float*)d_in[1];
  const float* b_qkv    = (const float*)d_in[2];
  const float* position = (const float*)d_in[3];
  const float* w_out    = (const float*)d_in[4];
  const float* b_out    = (const float*)d_in[5];
  float* y              = (float*)d_out;

  // Workspace layout (halfs): xh | q | k | v | o | whq | who  (~181 MB).
  const size_t SEG = (size_t)NBATCH * CIN * HPAD * WPAD;  // == q/k/v/o size
  _Float16* xh  = (_Float16*)d_ws;
  _Float16* qb  = xh + SEG;
  _Float16* kb  = qb + SEG;
  _Float16* vb  = kb + SEG;
  _Float16* obf = vb + SEG;
  _Float16* whq = obf + SEG;
  _Float16* who = whq + (size_t)(3 * HC) * CIN;

  // Pre-pass: pad+convert x, convert weights to fp16.
  const int nx = NBATCH * CIN * HPAD * WPAD;
  padcvt_kernel<<<dim3((nx + 255) / 256), dim3(256), 0, stream>>>(x, xh);
  cvt_kernel<<<dim3((3 * HC * CIN + 255) / 256), dim3(256), 0, stream>>>(
      w_qkv, whq, 3 * HC * CIN);
  cvt_kernel<<<dim3((CIN * HC + 255) / 256), dim3(256), 0, stream>>>(
      w_out, who, CIN * HC);

  dim3 blk(128);
  qkv_kernel<<<dim3(NBATCH * NWIN, 6), blk, 0, stream>>>(xh, whq, b_qkv,
                                                         position, qb, kb, vb);
  attn_kernel<<<dim3(NBATCH * HEADS * NWIN), blk, 0, stream>>>(qb, kb, vb, obf);
  out_kernel<<<dim3(NBATCH * NWIN, 2), blk, 0, stream>>>(obf, who, b_out, y);
}
